// GraphEncoder_44109314130281
// MI455X (gfx1250) — matrix operand order repaired and checked by reading, not compiled, not run on hardware
//
#include <hip/hip_runtime.h>

static constexpr int HD  = 128;   // hidden dim
static constexpr int GG  = 64;    // graphs in batch
static constexpr int LL  = 3;     // layers
static constexpr int NPB = 128;   // nodes per segment-reduce block
static constexpr int MBLK = 64;   // GEMM rows per block
static constexpr int LDA = 136;   // f16 per A row in LDS (128 data + 8 pad)
static constexpr int WPN = 40;    // f16 per (kt,n) B-fragment group (32 data + 8 pad)

typedef __attribute__((ext_vector_type(4)))  _Float16 v4h;
typedef __attribute__((ext_vector_type(8)))  _Float16 v8h;
typedef __attribute__((ext_vector_type(16))) _Float16 v16h;
typedef __attribute__((ext_vector_type(8)))  float    v8f;

// ---------- monotone float<->uint encoding for atomic max ----------
__device__ __forceinline__ unsigned enc_f(float f) {
    unsigned b = __float_as_uint(f);
    return (b & 0x80000000u) ? ~b : (b | 0x80000000u);
}
__device__ __forceinline__ float dec_f(unsigned u) {
    unsigned b = (u & 0x80000000u) ? (u ^ 0x80000000u) : ~u;
    return __uint_as_float(b);
}

// ---------------- setup kernels ----------------
__global__ void k_init(float* deg, float* counts, float* pool_sum,
                       unsigned* pool_max, int N) {
    int i = blockIdx.x * blockDim.x + threadIdx.x;
    if (i < N) deg[i] = 1.0f;                 // self-loop degree contribution
    if (i < GG) counts[i] = 0.0f;
    if (i < GG * HD) { pool_sum[i] = 0.0f; pool_max[i] = 0u; }
}

__global__ void k_deg_counts(const int* __restrict__ dst, const int* __restrict__ batch,
                             float* deg, float* counts, int E, int N) {
    int i = blockIdx.x * blockDim.x + threadIdx.x;
    if (i < E) atomicAdd(&deg[dst[i]], 1.0f);
    if (i < N) atomicAdd(&counts[batch[i]], 1.0f);
}

__global__ void k_dinv(float* deg, int N) {
    int i = blockIdx.x * blockDim.x + threadIdx.x;
    if (i < N) { float d = deg[i]; deg[i] = (d > 0.0f) ? rsqrtf(d) : 0.0f; }
}

// ---------------- input projection: h = relu(x @ w_in + b_in) ----------------
__global__ __launch_bounds__(256)
void k_in_proj(const float* __restrict__ x, const float* __restrict__ w_in,
               const float* __restrict__ b_in, float* __restrict__ h, int N) {
    __shared__ float wls[7 * HD];
    __shared__ float bls[HD];
    int tid = threadIdx.x;
    for (int i = tid; i < 7 * HD; i += 256) wls[i] = w_in[i];
    if (tid < HD) bls[tid] = b_in[tid];
    __syncthreads();
    size_t idx = (size_t)blockIdx.x * 256 + tid;
    if (idx >= (size_t)N * HD) return;
    int n = (int)(idx >> 7);
    int c = (int)(idx & 127);
    float acc = bls[c];
    const float* xr = x + (size_t)n * 7;
#pragma unroll
    for (int k = 0; k < 7; ++k) acc += xr[k] * wls[k * HD + c];
    h[idx] = fmaxf(acc, 0.0f);
}

// ---------------- WMMA GEMM: m = h(fp32->f16) @ W(fp32->f16), fp32 accum ----------------
// A is staged in LDS permuted into WMMA-fragment order so each lane's 16-f16
// fragment is 32 contiguous bytes (2x ds_load_b128). W is staged fragment-major
// per (kt, n, half-wave) with 8-f16 padding per group for bank spread.
__global__ __launch_bounds__(256)
void k_gemm_wmma(const float* __restrict__ hsrc, const float* __restrict__ Wsrc,
                 float* __restrict__ mout, int N) {
    __shared__ _Float16 Wb[4 * HD * WPN];   // 40960 B
    __shared__ _Float16 Al[MBLK * LDA];     // 17408 B
    int tid = threadIdx.x;

    // ---- stage W: Wb[(kt*128+n)*WPN + hi*16 + jj] = W[kt*32+hi*16+jj][n]
    for (int i = tid * 4; i < HD * HD; i += 256 * 4) {
        float4 w4 = *(const float4*)(Wsrc + i);
        int k  = i >> 7;
        int n  = i & 127;
        int kt = k >> 5, rem = k & 31;
        int hi = rem >> 4, jj = rem & 15;
        int base = (kt * HD) * WPN + hi * 16 + jj;
        Wb[base + (n + 0) * WPN] = (_Float16)w4.x;
        Wb[base + (n + 1) * WPN] = (_Float16)w4.y;
        Wb[base + (n + 2) * WPN] = (_Float16)w4.z;
        Wb[base + (n + 3) * WPN] = (_Float16)w4.w;
    }

    // ---- stage A permuted: column k -> p = kt*32 + hi*16 + sec*8 + j
    // (fragment (m,hi,kt) then occupies Al[m*LDA + kt*32 + hi*16 .. +16))
    int row0 = blockIdx.x * MBLK;
    for (int i = tid * 4; i < MBLK * HD; i += 256 * 4) {
        int r = i >> 7;
        int c = i & 127;
        int row = row0 + r;
        float4 v4 = {0.f, 0.f, 0.f, 0.f};
        if (row < N) v4 = *(const float4*)(hsrc + (size_t)row * HD + c);
        int kt = c >> 5, rem = c & 31;
        int hi = (rem >> 4) & 1, sec = (rem >> 3) & 1, j = rem & 7;
        int p = kt * 32 + hi * 16 + sec * 8 + j;
        v4h hv = {(_Float16)v4.x, (_Float16)v4.y, (_Float16)v4.z, (_Float16)v4.w};
        *(v4h*)&Al[r * LDA + p] = hv;   // 8B aligned packed store
    }
    __syncthreads();

    int lane = tid & 31;
    int wv   = tid >> 5;        // wave id 0..7 -> 16-col slab
    int n0   = wv * 16;
    int nn   = lane & 15;
    int hi   = lane >> 4;       // half-wave select
    bool full = (row0 + MBLK <= N);

    // ---- B fragments: 2x 16B LDS loads each, held in registers
    v16h bf[4];
#pragma unroll
    for (int kt = 0; kt < 4; ++kt) {
        const _Float16* bp = &Wb[(kt * HD + n0 + nn) * WPN + hi * 16];
        v8h lo = *(const v8h*)bp;
        v8h h8 = *(const v8h*)(bp + 8);
        bf[kt] = __builtin_shufflevector(lo, h8, 0, 1, 2, 3, 4, 5, 6, 7,
                                                 8, 9, 10, 11, 12, 13, 14, 15);
    }

#pragma unroll
    for (int mt = 0; mt < 4; ++mt) {
        v8f c = {0.f, 0.f, 0.f, 0.f, 0.f, 0.f, 0.f, 0.f};
#pragma unroll
        for (int kt = 0; kt < 4; ++kt) {
            const _Float16* ap = &Al[(mt * 16 + nn) * LDA + kt * 32 + hi * 16];
            v8h lo = *(const v8h*)ap;
            v8h h8 = *(const v8h*)(ap + 8);
            v16h a = __builtin_shufflevector(lo, h8, 0, 1, 2, 3, 4, 5, 6, 7,
                                                     8, 9, 10, 11, 12, 13, 14, 15);
            c = __builtin_amdgcn_wmma_f32_16x16x32_f16(
                    false, a, false, bf[kt], (short)0, c, false, false);
        }
        // C layout: lane n=nn, VGPR r -> M = r + 8*hi
        float* outp = mout + (size_t)(row0 + mt * 16 + hi * 8) * HD + n0 + nn;
        if (full) {
#pragma unroll
            for (int r = 0; r < 8; ++r) outp[(size_t)r * HD] = c[r];
        } else {
#pragma unroll
            for (int r = 0; r < 8; ++r)
                if (row0 + mt * 16 + hi * 8 + r < N) outp[(size_t)r * HD] = c[r];
        }
    }
}

// ---------------- agg = m*dinv^2 + conv_b (self loop) ; zero layer stats ----------------
__global__ void k_agg_init(const float* __restrict__ m, const float* __restrict__ dinv,
                           const float* __restrict__ conv_b, float* __restrict__ agg,
                           float* meansum, float* varsum, int N) {
    size_t idx = (size_t)blockIdx.x * blockDim.x + threadIdx.x;
    if (idx < (size_t)GG * HD) { meansum[idx] = 0.0f; varsum[idx] = 0.0f; }
    if (idx >= (size_t)N * HD) return;
    int n = (int)(idx >> 7);
    int c = (int)(idx & 127);
    float di = dinv[n];
    agg[idx] = m[idx] * di * di + conv_b[c];
}

// ---------------- edge scatter: wave per edge, lane = 4 channels ----------------
__global__ void k_agg_edges(const float4* __restrict__ m4, const int* __restrict__ src,
                            const int* __restrict__ dst, const float* __restrict__ dinv,
                            float* __restrict__ agg, int E) {
    size_t t = (size_t)blockIdx.x * blockDim.x + threadIdx.x;
    int e = (int)(t >> 5);
    int lane = (int)(t & 31);
    if (e >= E) return;
    int s = src[e], d = dst[e];
    float w = dinv[s] * dinv[d];
    float4 v = m4[(size_t)s * 32 + lane];
    float* out = agg + (size_t)d * HD + lane * 4;
    atomicAdd(out + 0, v.x * w);
    atomicAdd(out + 1, v.y * w);
    atomicAdd(out + 2, v.z * w);
    atomicAdd(out + 3, v.w * w);
}

// ---------------- segment sum over sorted batch (one flush per graph change) ----------------
__global__ __launch_bounds__(128)
void k_seg_sum(const float* __restrict__ v, const int* __restrict__ batch,
               float* __restrict__ out, int N) {
    __shared__ int bl[NPB];
    int c = threadIdx.x;
    int nstart = blockIdx.x * NPB;
    { int n = nstart + c; bl[c] = (n < N) ? batch[n] : -1; }
    __syncthreads();
    float acc = 0.0f;
    int curg = bl[0];
    for (int i = 0; i < NPB; ++i) {
        int g = bl[i];
        if (g < 0) break;
        if (g != curg) { atomicAdd(&out[(size_t)curg * HD + c], acc); acc = 0.0f; curg = g; }
        acc += v[(size_t)(nstart + i) * HD + c];
    }
    if (curg >= 0) atomicAdd(&out[(size_t)curg * HD + c], acc);
}

__global__ void k_mean_div(const float* meansum, const float* counts, float* mean) {
    int j = blockIdx.x * blockDim.x + threadIdx.x;
    if (j >= GG * HD) return;
    mean[j] = meansum[j] / fmaxf(counts[j >> 7], 1.0f);
}

__global__ __launch_bounds__(128)
void k_seg_var(const float* __restrict__ agg, const int* __restrict__ batch,
               const float* __restrict__ mean, const float* __restrict__ alpha,
               float* __restrict__ varsum, int N) {
    __shared__ int bl[NPB];
    int c = threadIdx.x;
    int nstart = blockIdx.x * NPB;
    { int n = nstart + c; bl[c] = (n < N) ? batch[n] : -1; }
    __syncthreads();
    float al = alpha[c];
    float acc = 0.0f;
    int curg = bl[0];
    for (int i = 0; i < NPB; ++i) {
        int g = bl[i];
        if (g < 0) break;
        if (g != curg) { atomicAdd(&varsum[(size_t)curg * HD + c], acc); acc = 0.0f; curg = g; }
        float cv = agg[(size_t)(nstart + i) * HD + c] - al * mean[(size_t)g * HD + c];
        acc += cv * cv;
    }
    if (curg >= 0) atomicAdd(&varsum[(size_t)curg * HD + c], acc);
}

__global__ void k_rstd(const float* varsum, const float* counts, float* rstd) {
    int j = blockIdx.x * blockDim.x + threadIdx.x;
    if (j >= GG * HD) return;
    rstd[j] = rsqrtf(varsum[j] / fmaxf(counts[j >> 7], 1.0f) + 1e-5f);
}

// ---------------- h = relu(w*(agg - a*mean)*rstd + b) + h  (residual, in place) ----------------
__global__ void k_update(const float* __restrict__ agg, const int* __restrict__ batch,
                         const float* __restrict__ mean, const float* __restrict__ rstd,
                         const float* __restrict__ nw, const float* __restrict__ nb,
                         const float* __restrict__ na, float* __restrict__ h, int N) {
    size_t idx = (size_t)blockIdx.x * blockDim.x + threadIdx.x;
    if (idx >= (size_t)N * HD) return;
    int n = (int)(idx >> 7);
    int c = (int)(idx & 127);
    int g = batch[n];
    float cv = agg[idx] - na[c] * mean[(size_t)g * HD + c];
    float hn = nw[c] * cv * rstd[(size_t)g * HD + c] + nb[c];
    h[idx] = fmaxf(hn, 0.0f) + h[idx];
}

// ---------------- pooling: per-graph sum + max over sorted batch ----------------
__global__ __launch_bounds__(128)
void k_pool(const float* __restrict__ h, const int* __restrict__ batch,
            float* __restrict__ psum, unsigned* __restrict__ pmax, int N) {
    __shared__ int bl[NPB];
    int c = threadIdx.x;
    int nstart = blockIdx.x * NPB;
    { int n = nstart + c; bl[c] = (n < N) ? batch[n] : -1; }
    __syncthreads();
    float acc = 0.0f;
    unsigned mx = 0u;
    int curg = bl[0];
    for (int i = 0; i < NPB; ++i) {
        int g = bl[i];
        if (g < 0) break;
        if (g != curg) {
            atomicAdd(&psum[(size_t)curg * HD + c], acc);
            atomicMax(&pmax[(size_t)curg * HD + c], mx);
            acc = 0.0f; mx = 0u; curg = g;
        }
        float hv = h[(size_t)(nstart + i) * HD + c];
        acc += hv;
        unsigned e = enc_f(hv);
        mx = (e > mx) ? e : mx;
    }
    if (curg >= 0) {
        atomicAdd(&psum[(size_t)curg * HD + c], acc);
        atomicMax(&pmax[(size_t)curg * HD + c], mx);
    }
}

__global__ void k_emb(const float* psum, const unsigned* pmax, const float* counts,
                      float* emb) {
    int j = blockIdx.x * blockDim.x + threadIdx.x;
    if (j >= GG * HD) return;
    int g = j >> 7, c = j & 127;
    float cnt = fmaxf(counts[g], 1.0f);
    emb[(size_t)g * (2 * HD) + c]      = psum[j] / cnt;
    emb[(size_t)g * (2 * HD) + HD + c] = dec_f(pmax[j]);
}

// =====================================================================
extern "C" void kernel_launch(void* const* d_in, const int* in_sizes, int n_in,
                              void* d_out, int out_size, void* d_ws, size_t ws_size,
                              hipStream_t stream) {
    (void)n_in; (void)out_size; (void)ws_size;
    const float* x        = (const float*)d_in[0];
    const int*   ei       = (const int*)d_in[1];
    const int*   batch    = (const int*)d_in[2];
    const float* w_in     = (const float*)d_in[3];
    const float* b_in     = (const float*)d_in[4];
    const float* conv_w   = (const float*)d_in[5];
    const float* conv_b   = (const float*)d_in[6];
    const float* norm_w   = (const float*)d_in[7];
    const float* norm_b   = (const float*)d_in[8];
    const float* norm_a   = (const float*)d_in[9];

    const int N = in_sizes[0] / 7;
    const int E = in_sizes[1] / 2;
    const int* src = ei;
    const int* dst = ei + E;

    float* h   = (float*)d_out;                       // [N, 128] final node features
    float* emb = h + (size_t)N * HD;                  // [64, 256] graph embeddings

    // ---- workspace carve-up (256B aligned) ----
    char* wp = (char*)d_ws;
    auto alloc = [&](size_t bytes) -> void* {
        void* p = (void*)wp;
        wp += (bytes + 255) & ~(size_t)255;
        return p;
    };
    float*    dinv    = (float*)alloc((size_t)N * sizeof(float));     // deg -> dinv
    float*    counts  = (float*)alloc(GG * sizeof(float));
    float*    mbuf    = (float*)alloc((size_t)N * HD * sizeof(float));
    float*    agg     = (float*)alloc((size_t)N * HD * sizeof(float));
    float*    meansum = (float*)alloc(GG * HD * sizeof(float));
    float*    mean    = (float*)alloc(GG * HD * sizeof(float));
    float*    varsum  = (float*)alloc(GG * HD * sizeof(float));
    float*    rstd    = (float*)alloc(GG * HD * sizeof(float));
    float*    psum    = (float*)alloc(GG * HD * sizeof(float));
    unsigned* pmax    = (unsigned*)alloc(GG * HD * sizeof(unsigned));

    const int TB = 256;
    const int gN   = (N + TB - 1) / TB;
    const int gE   = (E + TB - 1) / TB;
    const int gNH  = (int)(((size_t)N * HD + TB - 1) / TB);
    const int gGH  = (GG * HD + TB - 1) / TB;
    const int gSeg = (N + NPB - 1) / NPB;
    const int gEdg = (int)(((size_t)E * 32 + TB - 1) / TB);
    const int gMM  = (N + MBLK - 1) / MBLK;

    // ---- graph structure (shared across layers) ----
    k_init<<<gN, TB, 0, stream>>>(dinv, counts, psum, pmax, N);
    k_deg_counts<<<gE, TB, 0, stream>>>(dst, batch, dinv, counts, E, N);
    k_dinv<<<gN, TB, 0, stream>>>(dinv, N);

    // ---- input projection ----
    k_in_proj<<<gNH, TB, 0, stream>>>(x, w_in, b_in, h, N);

    // ---- layers ----
    for (int l = 0; l < LL; ++l) {
        k_gemm_wmma<<<gMM, TB, 0, stream>>>(h, conv_w + (size_t)l * HD * HD, mbuf, N);
        k_agg_init<<<gNH, TB, 0, stream>>>(mbuf, dinv, conv_b + (size_t)l * HD,
                                           agg, meansum, varsum, N);
        k_agg_edges<<<gEdg, TB, 0, stream>>>((const float4*)mbuf, src, dst, dinv, agg, E);
        k_seg_sum<<<gSeg, NPB, 0, stream>>>(agg, batch, meansum, N);
        k_mean_div<<<gGH, TB, 0, stream>>>(meansum, counts, mean);
        k_seg_var<<<gSeg, NPB, 0, stream>>>(agg, batch, mean, norm_a + (size_t)l * HD,
                                            varsum, N);
        k_rstd<<<gGH, TB, 0, stream>>>(varsum, counts, rstd);
        k_update<<<gNH, TB, 0, stream>>>(agg, batch, mean, rstd,
                                         norm_w + (size_t)l * HD, norm_b + (size_t)l * HD,
                                         norm_a + (size_t)l * HD, h, N);
    }

    // ---- readout ----
    k_pool<<<gSeg, NPB, 0, stream>>>(h, batch, psum, pmax, N);
    k_emb<<<gGH, TB, 0, stream>>>(psum, pmax, counts, emb);
}
